// ImaginariumModel_69260642615610
// MI455X (gfx1250) — compile-verified, matrix operand order checked by reading, not run on hardware
//
#include <hip/hip_runtime.h>
#include <hip/hip_bf16.h>

// ---------------------------------------------------------------------------
// Problem sizes (fixed by the reference): B=64, C=256, W=256, E=2048, D=2048
// ---------------------------------------------------------------------------
#define BB 64
#define CC 256
#define WW 256
#define EE 2048
#define DD 2048
#define EPSV 1e-8f

typedef __attribute__((ext_vector_type(16))) __bf16 v16bf;
typedef __attribute__((ext_vector_type(8)))  __bf16 v8bf;
typedef __attribute__((ext_vector_type(4)))  __bf16 bf16x4;
typedef __attribute__((ext_vector_type(8)))  float  v8f;

__device__ __forceinline__ v8f wmma_bf16(v16bf a, v16bf b, v8f c) {
    // D = A(16x32 bf16) * B(32x16 bf16) + C(16x16 f32)
    return __builtin_amdgcn_wmma_f32_16x16x32_bf16(
        false, a, false, b, (short)0, c, false, false);
}

__device__ __forceinline__ v16bf make_frag(v8bf lo, v8bf hi) {
    v16bf r;
#pragma unroll
    for (int i = 0; i < 8; ++i) { r[i] = lo[i]; r[i + 8] = hi[i]; }
    return r;
}

// Per-lane bf16 fragment load (matches 16-bit A/B VGPR layout):
// lanes 0-15 take K=[0,8)+[16,24), lanes 16-31 take K=[8,16)+[24,32)
// relative to k0. `base` must already include +khalf*8.
__device__ __forceinline__ v16bf load_frag(const __bf16* base) {
    v8bf lo = *(const v8bf*)(base);
    v8bf hi = *(const v8bf*)(base + 16);
    return make_frag(lo, hi);
}

__device__ __forceinline__ v8f zero_v8f() {
    v8f z;
#pragma unroll
    for (int i = 0; i < 8; ++i) z[i] = 0.0f;
    return z;
}

// ---------------------------------------------------------------------------
// Kernel 1: fp32 -> bf16 weight conversion (vectorized, one float4 per thread)
// ---------------------------------------------------------------------------
__global__ void cvt_f32_to_bf16(const float* __restrict__ in,
                                __bf16* __restrict__ out, int n4) {
    int i = blockIdx.x * blockDim.x + threadIdx.x;
    if (i < n4) {
        float4 v = ((const float4*)in)[i];
        bf16x4 o;
        o[0] = (__bf16)v.x; o[1] = (__bf16)v.y;
        o[2] = (__bf16)v.z; o[3] = (__bf16)v.w;
        ((bf16x4*)out)[i] = o;
    }
}

// ---------------------------------------------------------------------------
// Kernel 2: per-batch argmax over y[,C] + gather leader img row -> bf16
// ---------------------------------------------------------------------------
__global__ void argmax_gather(const float* __restrict__ y,
                              const float* __restrict__ x_img,
                              __bf16* __restrict__ leader_in) {
    __shared__ float sv[CC];
    __shared__ int   si[CC];
    const int b = blockIdx.x;
    const int t = threadIdx.x;

    sv[t] = y[b * CC + t];
    si[t] = t;
    __syncthreads();
#pragma unroll
    for (int s = CC / 2; s > 0; s >>= 1) {
        if (t < s) {
            float av = sv[t], bv = sv[t + s];
            int   ai = si[t], bi = si[t + s];
            // lexicographic: max value, first occurrence on ties
            if (bv > av || (bv == av && bi < ai)) { sv[t] = bv; si[t] = bi; }
        }
        __syncthreads();
    }
    const int idx = si[0];
    const float* src = x_img + ((size_t)b * CC + idx) * EE;
#pragma unroll
    for (int e = t; e < EE; e += CC)
        leader_in[b * EE + e] = (__bf16)src[e];
}

// ---------------------------------------------------------------------------
// Kernel 3: leader GEMM  leader[b,d] = sum_e lin[b,e]*W_img[d,e] + b_img[d]
//           M=64(batch) x N=2048(D) x K=2048(E), one 16x16 tile per wave.
// ---------------------------------------------------------------------------
__global__ void leader_gemm(const __bf16* __restrict__ A,   // [64][2048]
                            const __bf16* __restrict__ Wb,  // [2048][2048]
                            const float*  __restrict__ bias,
                            float* __restrict__ leader) {
    const int lane  = threadIdx.x & 31;
    const int wave  = threadIdx.x >> 5;
    const int tid   = blockIdx.x * 8 + wave;   // 512 tiles
    const int mt    = tid & 3;                 // 4 M tiles
    const int nt    = tid >> 2;                // 128 N tiles
    const int ln15  = lane & 15;
    const int khalf = lane >> 4;

    const __bf16* arow = A  + (size_t)(mt * 16 + ln15) * EE + khalf * 8;
    const __bf16* brow = Wb + (size_t)(nt * 16 + ln15) * EE + khalf * 8;

    v8f acc = zero_v8f();
#pragma unroll 4
    for (int k0 = 0; k0 < EE; k0 += 32) {
        v16bf a = load_frag(arow + k0);
        v16bf b = load_frag(brow + k0);
        acc = wmma_bf16(a, b, acc);
    }
    const int d = nt * 16 + ln15;
    const float bv = bias[d];
#pragma unroll
    for (int r = 0; r < 8; ++r) {
        const int m = mt * 16 + khalf * 8 + r;
        leader[(size_t)m * DD + d] = acc[r] + bv;
    }
}

// ---------------------------------------------------------------------------
// Kernel 4: leader L2 norm per batch
// ---------------------------------------------------------------------------
__global__ void leader_norm(const float* __restrict__ leader,
                            float* __restrict__ lnorm) {
    __shared__ float red[256];
    const int b = blockIdx.x, t = threadIdx.x;
    float s = 0.0f;
#pragma unroll
    for (int d = t; d < DD; d += 256) {
        float v = leader[(size_t)b * DD + d];
        s += v * v;
    }
    red[t] = s;
    __syncthreads();
#pragma unroll
    for (int k = 128; k > 0; k >>= 1) {
        if (t < k) red[t] += red[t + k];
        __syncthreads();
    }
    if (t == 0) lnorm[b] = sqrtf(red[0]);
}

// ---------------------------------------------------------------------------
// Kernel 5: fused main GEMM + cosine reduction.
//   One workgroup per (batch b, 64-word tile). A panel (64 x 2048 bf16) is
//   converted into LDS once; B (bf16 W_words) streams from L2. Epilogue
//   accumulates dot(xt,leader) and sumsq(xt) over D without storing xt.
// ---------------------------------------------------------------------------
#define WTILE 64
#define APITCH (EE + 8)  // +16B row pad to dodge LDS bank aliasing

__global__ void __launch_bounds__(256, 1)
fused_words(const float*  __restrict__ x_txt,   // [B][W][E] f32
            const __bf16* __restrict__ Wb,      // [D][E] bf16
            const float*  __restrict__ bias,    // [D]
            const float*  __restrict__ leader,  // [B][D]
            float* __restrict__ dotg,           // [B][W]
            float* __restrict__ sqg) {          // [B][W]
    extern __shared__ __bf16 aLds[];            // [WTILE][APITCH]
    __shared__ float dotS[WTILE];
    __shared__ float sqS[WTILE];

    const int b  = blockIdx.x >> 2;
    const int w0 = (blockIdx.x & 3) * WTILE;
    const int t  = threadIdx.x;
    if (t < WTILE) { dotS[t] = 0.0f; sqS[t] = 0.0f; }

    // ---- stage 1: x_txt panel -> bf16 in LDS (read fp32 from HBM once) ----
    const float* xr = x_txt + ((size_t)b * WW + w0) * EE;
#pragma unroll 4
    for (int i = t; i < WTILE * (EE / 4); i += 256) {
        const int row = i >> 9;          // EE/4 = 512 float4 per row
        const int q   = i & 511;
        float4 v = ((const float4*)(xr + (size_t)row * EE))[q];
        bf16x4 o;
        o[0] = (__bf16)v.x; o[1] = (__bf16)v.y;
        o[2] = (__bf16)v.z; o[3] = (__bf16)v.w;
        *(bf16x4*)(aLds + row * APITCH + q * 4) = o;
    }
    __syncthreads();

    // ---- stage 2: WMMA over all D tiles, fused reduction ----
    const int lane  = t & 31;
    const int wave  = t >> 5;
    const int m_sub = wave >> 1;    // 4 waves across M (rows of words)
    const int n_sub = wave & 1;     // 2 waves across N (D columns)
    const int ln15  = lane & 15;
    const int khalf = lane >> 4;

    const __bf16* arow = aLds + (m_sub * 16 + ln15) * APITCH + khalf * 8;

    float dotp[8], sqp[8];
#pragma unroll
    for (int r = 0; r < 8; ++r) { dotp[r] = 0.0f; sqp[r] = 0.0f; }

    for (int dt = 0; dt < DD / 64; ++dt) {
        const int d0 = dt * 64 + n_sub * 32 + ln15;   // lane's D column, frag 0
        const __bf16* brow0 = Wb + (size_t)d0        * EE + khalf * 8;
        const __bf16* brow1 = Wb + (size_t)(d0 + 16) * EE + khalf * 8;

        v8f acc0 = zero_v8f();
        v8f acc1 = zero_v8f();
#pragma unroll 4
        for (int k0 = 0; k0 < EE; k0 += 32) {
            v16bf a  = load_frag(arow + k0);     // ds_read_b128 x2
            v16bf b0 = load_frag(brow0 + k0);    // global_load_b128 x2 (L2 hot)
            v16bf b1 = load_frag(brow1 + k0);
            acc0 = wmma_bf16(a, b0, acc0);
            acc1 = wmma_bf16(a, b1, acc1);
        }
        const float lv0 = leader[(size_t)b * DD + d0];
        const float lv1 = leader[(size_t)b * DD + d0 + 16];
        const float bb0 = bias[d0];
        const float bb1 = bias[d0 + 16];
#pragma unroll
        for (int r = 0; r < 8; ++r) {
            float v0 = acc0[r] + bb0;
            float v1 = acc1[r] + bb1;
            dotp[r] += v0 * lv0 + v1 * lv1;
            sqp[r]  += v0 * v0 + v1 * v1;
        }
    }

    // ---- stage 3: cross-lane reduction over the D dimension ----
#pragma unroll
    for (int r = 0; r < 8; ++r) {
        const int m = m_sub * 16 + khalf * 8 + r;   // word row (C-layout)
        atomicAdd(&dotS[m], dotp[r]);               // ds_add_f32
        atomicAdd(&sqS[m],  sqp[r]);
    }
    __syncthreads();
    if (t < WTILE) {
        dotg[(size_t)b * WW + w0 + t] = dotS[t];
        sqg [(size_t)b * WW + w0 + t] = sqS[t];
    }
}

// ---------------------------------------------------------------------------
// Kernel 6: cosine logits + softmax over W=256 per batch
// ---------------------------------------------------------------------------
__global__ void cos_softmax(const float* __restrict__ dotg,
                            const float* __restrict__ sqg,
                            const float* __restrict__ lnorm,
                            float* __restrict__ out) {
    __shared__ float red[WW];
    const int b = blockIdx.x, t = threadIdx.x;
    const float denom = fmaxf(lnorm[b] * sqrtf(sqg[b * WW + t]), EPSV);
    const float logit = dotg[b * WW + t] / denom;

    red[t] = logit;
    __syncthreads();
#pragma unroll
    for (int k = WW / 2; k > 0; k >>= 1) {
        if (t < k) red[t] = fmaxf(red[t], red[t + k]);
        __syncthreads();
    }
    const float mx = red[0];
    __syncthreads();
    const float e = __expf(logit - mx);
    red[t] = e;
    __syncthreads();
#pragma unroll
    for (int k = WW / 2; k > 0; k >>= 1) {
        if (t < k) red[t] += red[t + k];
        __syncthreads();
    }
    out[b * WW + t] = e / red[0];
}

// ---------------------------------------------------------------------------
// Host-side launch
// ---------------------------------------------------------------------------
extern "C" void kernel_launch(void* const* d_in, const int* in_sizes, int n_in,
                              void* d_out, int out_size, void* d_ws, size_t ws_size,
                              hipStream_t stream) {
    const float* x_img   = (const float*)d_in[0];  // [B,C,E]
    const float* x_txt   = (const float*)d_in[1];  // [B,W,E]
    const float* y       = (const float*)d_in[2];  // [B,C]
    const float* W_img1  = (const float*)d_in[3];  // [D,E]
    const float* b_img1  = (const float*)d_in[4];  // [D]
    const float* W_wrd1  = (const float*)d_in[5];  // [D,E]
    const float* b_wrd1  = (const float*)d_in[6];  // [D]
    float* out = (float*)d_out;

    // workspace layout
    char* ws = (char*)d_ws;
    __bf16* Wwords_b  = (__bf16*)(ws);                       //  8,388,608 B
    __bf16* Wimg_b    = (__bf16*)(ws + 8388608);             //  8,388,608 B
    __bf16* leader_in = (__bf16*)(ws + 16777216);            //    262,144 B
    float*  leader    = (float*) (ws + 17039360);            //    524,288 B
    float*  lnorm     = (float*) (ws + 17563648);            //        256 B
    float*  dotg      = (float*) (ws + 17563904);            //     65,536 B
    float*  sqg       = (float*) (ws + 17629440);            //     65,536 B

    const int n4 = (DD * EE) / 4;  // 1,048,576 float4 per weight matrix
    cvt_f32_to_bf16<<<(n4 + 255) / 256, 256, 0, stream>>>(W_wrd1, Wwords_b, n4);
    cvt_f32_to_bf16<<<(n4 + 255) / 256, 256, 0, stream>>>(W_img1, Wimg_b, n4);

    argmax_gather<<<BB, CC, 0, stream>>>(y, x_img, leader_in);

    leader_gemm<<<64, 256, 0, stream>>>(leader_in, Wimg_b, b_img1, leader);
    leader_norm<<<BB, 256, 0, stream>>>(leader, lnorm);

    const size_t smem = (size_t)WTILE * APITCH * sizeof(__bf16); // 263,168 B
    fused_words<<<BB * (WW / WTILE), 256, smem, stream>>>(
        x_txt, Wwords_b, b_wrd1, leader, dotg, sqg);

    cos_softmax<<<BB, WW, 0, stream>>>(dotg, sqg, lnorm, out);
}
